// MicroTransformer_78099685310509
// MI455X (gfx1250) — compile-verified, matrix operand order checked by reading, not run on hardware
//
#include <hip/hip_runtime.h>
#include <hip/hip_bf16.h>

// ---------------------------------------------------------------------------
// MicroTransformer forward for gfx1250 (MI455X).
// bf16 WMMA GEMMs (v_wmma_f32_16x16x32_bf16); GEMM tile staging via the
// Tensor Data Mover (tensor_load_to_lds + s_wait_tensorcnt) with async-LDS
// (global_load_async_to_lds_b128 + s_wait_asynccnt) as fallback.
// fp32 residual stream, bf16 activations/weights.
// ---------------------------------------------------------------------------

#ifndef USE_TDM
#define USE_TDM 1
#endif
#ifndef USE_ASYNC_LDS
#define USE_ASYNC_LDS 1
#endif

#if USE_TDM && __has_builtin(__builtin_amdgcn_tensor_load_to_lds)
#define HAVE_TDM 1
#else
#define HAVE_TDM 0
#endif

// amdgpu-toolchain (therock headers) uses the 6-arg tensor builtin; ROCm 7.2
// uses the 5-arg one. The TDM header only ships with the former.
#if __has_include(<hip/amd_detail/amd_gfx1250_TDM.h>)
#define TDM_SIX_ARGS 1
#else
#define TDM_SIX_ARGS 0
#endif

typedef __bf16 bf16;
typedef __attribute__((ext_vector_type(16))) __bf16   v16bf;
typedef __attribute__((ext_vector_type(8)))  float    v8f;
typedef __attribute__((ext_vector_type(4)))  unsigned u32x4;
typedef __attribute__((ext_vector_type(8)))  int      i32x8;
typedef __attribute__((ext_vector_type(4)))  int      i32x4;

static __device__ __forceinline__ bf16 f2bf(float f) {
    unsigned u = __builtin_bit_cast(unsigned, f);
    u += 0x7FFFu + ((u >> 16) & 1u);                 // round-to-nearest-even
    unsigned short s = (unsigned short)(u >> 16);
    return __builtin_bit_cast(bf16, s);
}

// generic (shared) pointer -> 32-bit LDS byte address
static __device__ __forceinline__ unsigned lds_addr_u32(const void* p) {
    return (unsigned)(unsigned long long)
           (const __attribute__((address_space(3))) void*)p;
}

#if HAVE_TDM
// One-shot 2D tile DMA: 128 rows x 32 bf16 from row-major [*, K] matrix into
// LDS, padding +16B after each 64B row (matches LDS_STRIDE=40 layout).
// D# fields per CDNA5 ISA 8.3/8.4 (group0: count/lds/global/type=2;
// group1: data_size=2B, pad 4 DWORDs per 16 DWORDs, dims/strides).
static __device__ __forceinline__ void tdm_load_2d(const void* gaddr,
                                                   unsigned lds, int K) {
    unsigned long long ga = (unsigned long long)(size_t)gaddr;
    u32x4 g0;
    g0[0] = 1u;                                           // count=1, user D#
    g0[1] = lds;                                          // lds_addr
    g0[2] = (unsigned)ga;                                 // global_addr[31:0]
    g0[3] = (unsigned)((ga >> 32) & 0x01FFFFFFu) | (2u << 30); // [56:32]|type=2
    i32x8 g1;
    g1[0] = (int)((1u << 16)       // data_size = 2 bytes
                | (1u << 20)       // pad_enable
                | (3u << 22)       // pad_interval: 16 DWORDs (64B)
                | (3u << 25));     // pad_amount: 4 DWORDs (16B)
    g1[1] = (int)(((unsigned)K & 0xFFFFu) << 16);         // tensor_dim0 lo16
    g1[2] = (int)(((unsigned)K >> 16) | (128u << 16));    // dim0 hi | dim1 lo
    g1[3] = (int)(32u << 16);                             // dim1 hi=0 | tile_dim0=32
    g1[4] = 128;                                          // tile_dim1=128, tile_dim2=0
    g1[5] = K;                                            // tensor_dim0_stride lo32
    g1[6] = 0;                                            // stride hi | dim1_stride lo
    g1[7] = 0;
    i32x4 z4 = {};
#if TDM_SIX_ARGS
    i32x8 z8 = {};
    __builtin_amdgcn_tensor_load_to_lds(g0, g1, z4, z4, z8, 0);
#else
    __builtin_amdgcn_tensor_load_to_lds(g0, g1, z4, z4, 0);
#endif
}

static __device__ __forceinline__ void wait_tensor0() {
#if __has_builtin(__builtin_amdgcn_s_wait_tensorcnt)
    __builtin_amdgcn_s_wait_tensorcnt((short)0);
#else
    asm volatile("s_wait_tensorcnt 0x0" ::: "memory");
#endif
}

#elif USE_ASYNC_LDS
// per-lane 16B async copy: LDS[lds] = MEM[gptr]  (tracked by ASYNCcnt)
static __device__ __forceinline__ void async_ld_b128(const void* gptr, unsigned lds) {
    asm volatile("global_load_async_to_lds_b128 %0, %1, off"
                 :: "v"(lds), "v"((unsigned long long)(size_t)gptr)
                 : "memory");
}

static __device__ __forceinline__ void wait_async0() {
#if __has_builtin(__builtin_amdgcn_s_wait_asynccnt)
    __builtin_amdgcn_s_wait_asynccnt(0);
#else
    asm volatile("s_wait_asynccnt 0x0" ::: "memory");
#endif
}
#endif

// --------------------------- elementwise kernels ---------------------------

__global__ void cast_f32_to_bf16(const float* __restrict__ src,
                                 bf16* __restrict__ dst, size_t n) {
    size_t i = (size_t)blockIdx.x * blockDim.x + threadIdx.x;
    size_t stride = (size_t)gridDim.x * blockDim.x;
    for (; i < n; i += stride) dst[i] = f2bf(src[i]);
}

__global__ void embed_kernel(const int* __restrict__ ids,
                             const float* __restrict__ emb,
                             float* __restrict__ x, int D) {
    int t = blockIdx.x;
    int id = ids[t];
    for (int i = threadIdx.x; i < D; i += blockDim.x)
        x[(size_t)t * D + i] = emb[(size_t)id * D + i];
}

__global__ __launch_bounds__(256)
void rmsnorm_kernel(const float* __restrict__ x, const float* __restrict__ w,
                    bf16* __restrict__ h, int D) {
    int t = blockIdx.x;
    const float* xr = x + (size_t)t * D;
    __shared__ float red[256];
    float s = 0.f;
    for (int i = threadIdx.x; i < D; i += 256) { float v = xr[i]; s += v * v; }
    red[threadIdx.x] = s;
    __syncthreads();
    for (int o = 128; o > 0; o >>= 1) {
        if (threadIdx.x < o) red[threadIdx.x] += red[threadIdx.x + o];
        __syncthreads();
    }
    float scale = rsqrtf(red[0] / (float)D + 1e-6f);
    for (int i = threadIdx.x; i < D; i += 256)
        h[(size_t)t * D + i] = f2bf(xr[i] * scale * w[i]);
}

__global__ void swiglu_kernel(const float* __restrict__ g,
                              const float* __restrict__ u,
                              bf16* __restrict__ out, size_t n) {
    size_t i = (size_t)blockIdx.x * blockDim.x + threadIdx.x;
    size_t stride = (size_t)gridDim.x * blockDim.x;
    for (; i < n; i += stride) {
        float gv = g[i];
        out[i] = f2bf((gv / (1.f + __expf(-gv))) * u[i]);
    }
}

// ------------------------------- RoPE scatter ------------------------------
// qkv: [T, 3D]  ->  q/k/v: [B*H, S, 64] (q,k rotated)

__global__ __launch_bounds__(64)
void rope_kernel(const float* __restrict__ qkv, float* __restrict__ qb,
                 float* __restrict__ kb, float* __restrict__ vb,
                 int S, int H, int D) {
    int blk = blockIdx.x;                 // t*H + h
    int t = blk / H, h = blk % H;
    int s = t % S, b = t / S;
    int i = threadIdx.x;                  // 0..63 (head dim)
    const float* base = qkv + (size_t)t * (3 * D);
    float qv = base[h * 64 + i];
    float kv = base[D + h * 64 + i];
    float vv = base[2 * D + h * 64 + i];
    float qo = (i < 32) ? -base[h * 64 + i + 32] : base[h * 64 + i - 32];
    float ko = (i < 32) ? -base[D + h * 64 + i + 32] : base[D + h * 64 + i - 32];
    int j = i & 31;
    float inv = __powf(10000.0f, -(float)(2 * j) * (1.0f / 64.0f));
    float ang = (float)s * inv;
    float c = __cosf(ang), sn = __sinf(ang);
    size_t idx = ((size_t)(b * H + h) * S + s) * 64 + i;
    qb[idx] = qv * c + qo * sn;
    kb[idx] = kv * c + ko * sn;
    vb[idx] = vv;
}

// --------------------------- attention (online softmax) --------------------
// one wave32 per query row; lane owns head-dims {2*lane, 2*lane+1}
// writes o as bf16 in [T, D] layout (col = h*64+d), ready for out-proj GEMM.

__global__ __launch_bounds__(256)
void attn_kernel(const float* __restrict__ q, const float* __restrict__ k,
                 const float* __restrict__ v, bf16* __restrict__ o,
                 int S, int H) {
    int wave = threadIdx.x >> 5, lane = threadIdx.x & 31;
    int row = blockIdx.x * 8 + wave;      // (b*H+h)*S + qpos
    int bh = row / S, qpos = row % S;
    int b = bh / H, h = bh % H;
    int d0 = lane * 2;
    const float* qr = q + ((size_t)bh * S + qpos) * 64;
    const float* kb = k + (size_t)bh * S * 64;
    const float* vb = v + (size_t)bh * S * 64;
    float q0 = qr[d0], q1 = qr[d0 + 1];
    float m = -3.0e38f, l = 0.f, a0 = 0.f, a1 = 0.f;
    const float scale = 0.125f;           // 1/sqrt(64)
    for (int j = 0; j <= qpos; ++j) {
        const float* kr = kb + (size_t)j * 64;
        float p = q0 * kr[d0] + q1 * kr[d0 + 1];
        #pragma unroll
        for (int off = 16; off >= 1; off >>= 1) p += __shfl_xor(p, off, 32);
        float sc = p * scale;
        float mn = fmaxf(m, sc);
        float corr = __expf(m - mn);
        float w = __expf(sc - mn);
        const float* vr = vb + (size_t)j * 64;
        l = l * corr + w;
        a0 = a0 * corr + w * vr[d0];
        a1 = a1 * corr + w * vr[d0 + 1];
        m = mn;
    }
    size_t t = (size_t)b * S + qpos;
    bf16* orow = o + t * (size_t)(H * 64) + h * 64;
    orow[d0]     = f2bf(a0 / l);
    orow[d0 + 1] = f2bf(a1 / l);
}

// ------------------------------ WMMA GEMM ----------------------------------
// C[M,N] (fp32, optional +=residual-in-place) = A[M,K](bf16) * W[N,K]^T(bf16)
// 128x128 C tile / block, 8 wave32s; wave (w) owns M tiles {w&3, (w&3)+4} and
// N tiles {(w>>2)*4 .. +3} -> 8 accumulators, 8 WMMAs per 32-deep K slab from
// 6 fragment loads (2 A + 4 B, B shared across both M tiles).
// Tile staging: TDM (waves 0/1 each issue one tensor_load_to_lds per slab,
// TENSORcnt) or per-lane async-LDS copies (ASYNCcnt). Double-buffered.
// LDS rows padded to 40 bf16 (80B) to spread fragment reads across banks
// (TDM reproduces the padding via D# pad_interval/pad_amount).
// Fragment packing per ISA 7.12.2 (16-bit A 16x32): lane m (0-15) holds row m,
// K in {0..7,16..23}; lane m+16 holds K in {8..15,24..31}; B mirrored over N.

#define LDS_STRIDE    40
#define LDS_BUF_BYTES (128 * LDS_STRIDE * 2)   // 10240 B per buffer

static __device__ __forceinline__ v16bf frag_ld(const bf16* rowp, int lane) {
    int klo = (lane & 16) ? 8 : 0;
    union { uint4 u[2]; v16bf v; } f;
    f.u[0] = *(const uint4*)(rowp + klo);
    f.u[1] = *(const uint4*)(rowp + klo + 16);
    return f.v;
}

__global__ __launch_bounds__(256)
void gemm_bf16_kernel(const bf16* __restrict__ A, const bf16* __restrict__ W,
                      float* __restrict__ C, int M, int N, int K, int residual) {
    __shared__ bf16 As[2][128][LDS_STRIDE];
    __shared__ bf16 Ws[2][128][LDS_STRIDE];
    const int tid  = threadIdx.x;
    const int lane = tid & 31;
    const int wave = tid >> 5;
    const int mtb = wave & 3;             // M tiles {mtb, mtb+4}
    const int ntb = (wave >> 2) * 4;      // N tiles {ntb .. ntb+3}
    const int bm = blockIdx.y * 128;
    const int bn = blockIdx.x * 128;

    const unsigned as0 = lds_addr_u32(&As[0][0][0]);
    const unsigned ws0 = lds_addr_u32(&Ws[0][0][0]);

#if HAVE_TDM
    auto stage = [&](int k0, int buf) {
        if (wave == 0)
            tdm_load_2d(A + (size_t)bm * K + k0, as0 + buf * LDS_BUF_BYTES, K);
        else if (wave == 1)
            tdm_load_2d(W + (size_t)bn * K + k0, ws0 + buf * LDS_BUF_BYTES, K);
    };
    auto wait_stage = [&]() { if (wave < 2) wait_tensor0(); };
#else
    // staging map: 256 rows x 32 bf16 = 1024 x 16B chunks, 4 per thread;
    // LDS destinations precomputed as 32-bit offsets (buffer chosen by add,
    // no pointer arrays -> no v_movrels / per-stage addrspace null checks).
    const bf16* srcbase[4];
    unsigned    ldsoff[4];
    #pragma unroll
    for (int j = 0; j < 4; ++j) {
        int c = tid + 256 * j;
        int row = c >> 2, col = (c & 3) * 8;
        if (row < 128) {
            srcbase[j] = A + (size_t)(bm + row) * K + col;
            ldsoff[j]  = as0 + row * (LDS_STRIDE * 2) + col * 2;
        } else {
            srcbase[j] = W + (size_t)(bn + row - 128) * K + col;
            ldsoff[j]  = ws0 + (row - 128) * (LDS_STRIDE * 2) + col * 2;
        }
    }
    auto stage = [&](int k0, int buf) {
        #pragma unroll
        for (int j = 0; j < 4; ++j) {
#if USE_ASYNC_LDS
            async_ld_b128(srcbase[j] + k0, ldsoff[j] + buf * LDS_BUF_BYTES);
#else
            bf16* dst = (bf16*)((char*)&As[0][0][0] +
                                (ldsoff[j] - as0) + buf * LDS_BUF_BYTES);
            *(uint4*)dst = *(const uint4*)(srcbase[j] + k0);
#endif
        }
    };
    auto wait_stage = [&]() {
#if USE_ASYNC_LDS
        wait_async0();
#endif
    };
#endif

    v8f acc[2][4] = {};

    const int nslab = K / 32;
    stage(0, 0);
    for (int i = 0; i < nslab; ++i) {
        const int buf = i & 1;
        wait_stage();                     // my DMA/async writes for slab i landed
        __syncthreads();                  // all waves' slab-i data visible
        if (i + 1 < nslab) stage((i + 1) * 32, buf ^ 1);

        v16bf bf[4];
        #pragma unroll
        for (int n = 0; n < 4; ++n)
            bf[n] = frag_ld(&Ws[buf][(ntb + n) * 16 + (lane & 15)][0], lane);
        #pragma unroll
        for (int mi = 0; mi < 2; ++mi) {
            v16bf af = frag_ld(&As[buf][(mtb + 4 * mi) * 16 + (lane & 15)][0], lane);
            #pragma unroll
            for (int n = 0; n < 4; ++n)
                acc[mi][n] = __builtin_amdgcn_wmma_f32_16x16x32_bf16(
                                 false, af, false, bf[n], (short)0,
                                 acc[mi][n], false, false);
        }
        __syncthreads();                  // done reading buf before overwrite
    }

    // C fragment layout: VGPR r = row r (lanes 0-15) / row r+8 (lanes 16-31)
    #pragma unroll
    for (int mi = 0; mi < 2; ++mi) {
        const int r0 = bm + (mtb + 4 * mi) * 16 + ((lane & 16) ? 8 : 0);
        #pragma unroll
        for (int n = 0; n < 4; ++n) {
            const int col = bn + (ntb + n) * 16 + (lane & 15);
            #pragma unroll
            for (int r = 0; r < 8; ++r) {
                size_t idx = (size_t)(r0 + r) * N + col;
                float v = acc[mi][n][r];
                if (residual) v += C[idx];
                C[idx] = v;
            }
        }
    }
}

// ------------------------------- host driver -------------------------------

extern "C" void kernel_launch(void* const* d_in, const int* in_sizes, int n_in,
                              void* d_out, int out_size, void* d_ws, size_t ws_size,
                              hipStream_t stream) {
    (void)in_sizes; (void)n_in; (void)out_size; (void)ws_size;
    constexpr int B = 2, S = 2048, T = B * S, D = 1024, H = 16,
                  F = 4096, V = 32000, L = 8;

    const int*   ids      = (const int*)d_in[0];
    const float* emb      = (const float*)d_in[1];
    const float* qkv_w    = (const float*)d_in[2];
    const float* out_w    = (const float*)d_in[3];
    const float* w1       = (const float*)d_in[4];
    const float* w2       = (const float*)d_in[5];
    const float* w3       = (const float*)d_in[6];
    const float* attn_nw  = (const float*)d_in[7];
    const float* ffn_nw   = (const float*)d_in[8];
    const float* final_nw = (const float*)d_in[9];
    float* logits = (float*)d_out;

    // workspace bump allocator (256B aligned)
    char* p = (char*)d_ws;
    auto alloc = [&](size_t bytes) -> void* {
        void* r = (void*)p;
        p += (bytes + 255) & ~(size_t)255;
        return r;
    };
    const size_t EMB_N = (size_t)V * D;
    const size_t QKV_N = (size_t)L * 3 * D * D;
    const size_t OUT_N = (size_t)L * D * D;
    const size_t W1_N  = (size_t)L * F * D;   // w2, w3 same size

    bf16*  emb_b = (bf16*)alloc(EMB_N * 2);
    bf16*  qkv_b = (bf16*)alloc(QKV_N * 2);
    bf16*  out_b = (bf16*)alloc(OUT_N * 2);
    bf16*  w1_b  = (bf16*)alloc(W1_N * 2);
    bf16*  w2_b  = (bf16*)alloc(W1_N * 2);
    bf16*  w3_b  = (bf16*)alloc(W1_N * 2);
    float* x     = (float*)alloc((size_t)T * D * 4);
    bf16*  h     = (bf16*)alloc((size_t)T * D * 2);
    float* qkvb  = (float*)alloc((size_t)T * 3 * D * 4);
    float* qb    = (float*)alloc((size_t)T * D * 4);
    float* kb    = (float*)alloc((size_t)T * D * 4);
    float* vb    = (float*)alloc((size_t)T * D * 4);
    bf16*  ob    = (bf16*)alloc((size_t)T * D * 2);
    float* g     = (float*)alloc((size_t)T * F * 4);
    float* u     = (float*)alloc((size_t)T * F * 4);
    bf16*  gu    = (bf16*)alloc((size_t)T * F * 2);

    // 1) weights -> bf16
    cast_f32_to_bf16<<<4096, 256, 0, stream>>>(emb,   emb_b, EMB_N);
    cast_f32_to_bf16<<<4096, 256, 0, stream>>>(qkv_w, qkv_b, QKV_N);
    cast_f32_to_bf16<<<4096, 256, 0, stream>>>(out_w, out_b, OUT_N);
    cast_f32_to_bf16<<<4096, 256, 0, stream>>>(w1,    w1_b,  W1_N);
    cast_f32_to_bf16<<<4096, 256, 0, stream>>>(w2,    w2_b,  W1_N);
    cast_f32_to_bf16<<<4096, 256, 0, stream>>>(w3,    w3_b,  W1_N);

    // 2) embedding gather (fp32 residual stream)
    embed_kernel<<<T, 256, 0, stream>>>(ids, emb, x, D);

    auto gemm = [&](const bf16* Am, const bf16* Wm, float* Cm,
                    int M, int N, int K, int res) {
        gemm_bf16_kernel<<<dim3(N / 128, M / 128), 256, 0, stream>>>(
            Am, Wm, Cm, M, N, K, res);
    };

    // 3) transformer blocks
    for (int l = 0; l < L; ++l) {
        rmsnorm_kernel<<<T, 256, 0, stream>>>(x, attn_nw + (size_t)l * D, h, D);
        gemm(h, qkv_b + (size_t)l * 3 * D * D, qkvb, T, 3 * D, D, 0);
        rope_kernel<<<T * H, 64, 0, stream>>>(qkvb, qb, kb, vb, S, H, D);
        attn_kernel<<<(B * H * S) / 8, 256, 0, stream>>>(qb, kb, vb, ob, S, H);
        gemm(ob, out_b + (size_t)l * D * D, x, T, D, D, 1);

        rmsnorm_kernel<<<T, 256, 0, stream>>>(x, ffn_nw + (size_t)l * D, h, D);
        gemm(h, w1_b + (size_t)l * F * D, g, T, F, D, 0);
        gemm(h, w3_b + (size_t)l * F * D, u, T, F, D, 0);
        swiglu_kernel<<<4096, 256, 0, stream>>>(g, u, gu, (size_t)T * F);
        gemm(gu, w2_b + (size_t)l * D * F, x, T, D, F, 1);
    }

    // 4) final norm + tied lm_head
    rmsnorm_kernel<<<T, 256, 0, stream>>>(x, final_nw, h, D);
    gemm(h, emb_b, logits, T, V, D, 0);
}
